// gcn_non_local_55533927137412
// MI455X (gfx1250) — matrix operand order repaired and checked by reading, not compiled, not run on hardware
//
#include <hip/hip_runtime.h>
#include <hip/hip_bf16.h>
#include <stdint.h>

// Problem constants (match reference)
#define BB   16
#define DD   256
#define LL   3
#define NDLC 2
#define NNV  1000   // valid rows
#define NP   1024   // padded rows (multiple of 32 and 64)

typedef __bf16 bf16;
typedef __attribute__((ext_vector_type(16))) __bf16 v16bf;
typedef __attribute__((ext_vector_type(8)))  float  v8f;

// ---------------- helpers ----------------

__device__ __forceinline__ bf16 f2bf(float f) {
  unsigned u = __builtin_bit_cast(unsigned, f);
  unsigned r = u + 0x7fffu + ((u >> 16) & 1u);   // round-to-nearest-even
  unsigned short h = (unsigned short)(r >> 16);
  return __builtin_bit_cast(bf16, h);
}

// A-matrix 16x32 bf16 tile (ISA 7.12.2): lane holds row M=lane&15.
// lanes 0-15: K = {0..7, 16..23}; lanes 16-31: K = {8..15, 24..31}.
__device__ __forceinline__ v16bf load_a_tile(const bf16* __restrict__ base,
                                             int lda, int k0, int lane) {
  int m  = lane & 15;
  int kb = (lane >> 4) << 3;                    // 0 or 8
  const bf16* p = base + (size_t)m * lda + k0 + kb;
  union { v16bf v; uint4 q[2]; } u;
  u.q[0] = *(const uint4*)(p);                  // K = kb .. kb+7
  u.q[1] = *(const uint4*)(p + 16);             // K = 16+kb .. 16+kb+7
  return u.v;
}

// B-matrix 32x16 bf16 tile, sourced from row-major B^T (row n contiguous in K).
// lane holds column N=lane&15; lanes 0-15: K = {0..7, 16..23}; lanes 16-31: K = {8..15, 24..31}.
__device__ __forceinline__ v16bf load_b_tile(const bf16* __restrict__ baseT,
                                             int ldb, int k0, int lane) {
  int n  = lane & 15;
  int kb = (lane >> 4) << 3;                    // 0 or 8
  const bf16* p = baseT + (size_t)n * ldb + k0 + kb;
  union { v16bf v; uint4 q[2]; } u;
  u.q[0] = *(const uint4*)(p);
  u.q[1] = *(const uint4*)(p + 16);
  return u.v;
}

#define WMMA_BF16(a, b, c) \
  __builtin_amdgcn_wmma_f32_16x16x32_bf16(false, (a), false, (b), (short)0, (c), false, false)

// ---------------- generic WMMA GEMM, 32x64 strip per wave ----------------
// C[m][n] = sum_k A[m][k] * Bt[n][k]   (Bt is B transposed, row-major)
// One wave computes a 32(M) x 64(N) tile: 2 A-frags x 4 B-frags = 8 WMMAs
// per K-step of 32 -> 24 B/lane of fresh global traffic per WMMA.
// M multiple of 32, Nn multiple of 64, K multiple of 32 (all padded).
// Optional per-batch Bt (m_per_batch>0): batch = m0/m_per_batch (strips never
// cross a batch boundary since m_per_batch % 32 == 0).
// Epilogue: +bias[n], ReLU, +skip[m][n], store fp32 and/or bf16.
__global__ void __launch_bounds__(256)
gcn_wmma_gemm(const bf16* __restrict__ A, const bf16* __restrict__ Bt,
              float* __restrict__ Cf, bf16* __restrict__ Cb,
              int M, int Nn, int K, int lda, int ldb, int ldc,
              size_t bt_bstride, int m_per_batch,
              const float* __restrict__ bias, int relu,
              const float* __restrict__ skip)
{
  int wave = (int)((blockIdx.x * blockDim.x + threadIdx.x) >> 5);
  int lane = threadIdx.x & 31;
  int mt_count = M >> 5, nt_count = Nn >> 6;
  if (wave >= mt_count * nt_count) return;      // uniform per wave
  int mt = wave / nt_count, nt = wave - mt * nt_count;
  int m0 = mt << 5, n0 = nt << 6;

  const bf16* Ab0 = A + (size_t)m0 * lda;
  const bf16* Ab1 = Ab0 + (size_t)16 * lda;
  size_t boff = m_per_batch ? (size_t)(m0 / m_per_batch) * bt_bstride : 0;
  const bf16* Bb0 = Bt + boff + (size_t)n0 * ldb;
  const bf16* Bb1 = Bb0 + (size_t)16 * ldb;
  const bf16* Bb2 = Bb0 + (size_t)32 * ldb;
  const bf16* Bb3 = Bb0 + (size_t)48 * ldb;

  v8f acc[8] = {};                              // acc[fi*4+fj]
  for (int k0 = 0; k0 < K; k0 += 32) {
    // prefetch next K-block (speculative; OOB silently dropped)
    __builtin_prefetch(Ab0 + (size_t)(lane & 15) * lda + k0 + 32, 0, 1);
    __builtin_prefetch(Bb0 + (size_t)(lane & 15) * ldb + k0 + 32, 0, 1);
    v16bf a0 = load_a_tile(Ab0, lda, k0, lane);
    v16bf a1 = load_a_tile(Ab1, lda, k0, lane);
    v16bf b0 = load_b_tile(Bb0, ldb, k0, lane);
    v16bf b1 = load_b_tile(Bb1, ldb, k0, lane);
    v16bf b2 = load_b_tile(Bb2, ldb, k0, lane);
    v16bf b3 = load_b_tile(Bb3, ldb, k0, lane);
    acc[0] = WMMA_BF16(a0, b0, acc[0]);
    acc[1] = WMMA_BF16(a0, b1, acc[1]);
    acc[2] = WMMA_BF16(a0, b2, acc[2]);
    acc[3] = WMMA_BF16(a0, b3, acc[3]);
    acc[4] = WMMA_BF16(a1, b0, acc[4]);
    acc[5] = WMMA_BF16(a1, b1, acc[5]);
    acc[6] = WMMA_BF16(a1, b2, acc[6]);
    acc[7] = WMMA_BF16(a1, b3, acc[7]);
  }

  int n  = lane & 15;
  int mb = (lane >> 4) << 3;                    // 0 or 8
  #pragma unroll
  for (int fi = 0; fi < 2; ++fi) {
    #pragma unroll
    for (int fj = 0; fj < 4; ++fj) {
      v8f c = acc[fi * 4 + fj];
      int col = n0 + (fj << 4) + n;
      float bi = bias ? bias[col] : 0.0f;
      #pragma unroll
      for (int r = 0; r < 8; ++r) {
        int row = m0 + (fi << 4) + mb + r;
        size_t o = (size_t)row * ldc + (size_t)col;
        float v = c[r] + bi;
        if (relu) v = v > 0.f ? v : 0.f;
        if (skip) v += skip[o];
        if (Cf) Cf[o] = v;
        if (Cb) Cb[o] = f2bf(v);
      }
    }
  }
}

// ---------------- scores kernel, 32x64 strip per wave ----------------
// scores = Y * x0^T per batch; epilogue: exp, +1e-5 on diag, mask by
// (contacts + I), store fp32 unnormalized s, atomicAdd row sums.
// Tiles the padded 1024x1024 score matrix exactly (pad rows/cols are zero
// in Yb/Xb so loads are safe); stores lane-guarded to the valid 1000x1000.
__global__ void __launch_bounds__(256)
gcn_scores(const bf16* __restrict__ Yb, const bf16* __restrict__ Xb,
           const float* __restrict__ contacts,
           float* __restrict__ sf, float* __restrict__ rowsum)
{
  const int IT = NP / 32;                       // 32 row-strips
  const int JT = NP / 64;                       // 16 col-strips
  int wave = (int)((blockIdx.x * blockDim.x + threadIdx.x) >> 5);
  int lane = threadIdx.x & 31;
  int tiles_per_b = IT * JT;                    // 512
  int b = wave / tiles_per_b;
  if (b >= BB) return;                          // uniform per wave
  int t  = wave - b * tiles_per_b;
  int it = t / JT, jt = t - it * JT;
  int i0 = it << 5, j0 = jt << 6;

  const bf16* Ab0 = Yb + ((size_t)b * NP + i0) * DD;
  const bf16* Ab1 = Ab0 + (size_t)16 * DD;
  const bf16* Bb0 = Xb + ((size_t)b * NP + j0) * DD;  // B^T rows = x0 rows
  const bf16* Bb1 = Bb0 + (size_t)16 * DD;
  const bf16* Bb2 = Bb0 + (size_t)32 * DD;
  const bf16* Bb3 = Bb0 + (size_t)48 * DD;

  v8f acc[8] = {};
  for (int k0 = 0; k0 < DD; k0 += 32) {
    v16bf a0 = load_a_tile(Ab0, DD, k0, lane);
    v16bf a1 = load_a_tile(Ab1, DD, k0, lane);
    v16bf b0 = load_b_tile(Bb0, DD, k0, lane);
    v16bf b1 = load_b_tile(Bb1, DD, k0, lane);
    v16bf b2 = load_b_tile(Bb2, DD, k0, lane);
    v16bf b3 = load_b_tile(Bb3, DD, k0, lane);
    acc[0] = WMMA_BF16(a0, b0, acc[0]);
    acc[1] = WMMA_BF16(a0, b1, acc[1]);
    acc[2] = WMMA_BF16(a0, b2, acc[2]);
    acc[3] = WMMA_BF16(a0, b3, acc[3]);
    acc[4] = WMMA_BF16(a1, b0, acc[4]);
    acc[5] = WMMA_BF16(a1, b1, acc[5]);
    acc[6] = WMMA_BF16(a1, b2, acc[6]);
    acc[7] = WMMA_BF16(a1, b3, acc[7]);
  }

  int n  = lane & 15;
  int mb = (lane >> 4) << 3;
  #pragma unroll
  for (int fi = 0; fi < 2; ++fi) {
    #pragma unroll
    for (int fj = 0; fj < 4; ++fj) {
      v8f c = acc[fi * 4 + fj];
      #pragma unroll
      for (int r = 0; r < 8; ++r) {
        int i = i0 + (fi << 4) + mb + r;
        int j = j0 + (fj << 4) + n;
        if (i < NNV && j < NNV) {
          float diag = (i == j) ? 1.0f : 0.0f;
          float e = __expf(c[r]) + 1e-5f * diag;
          float a = contacts[((size_t)b * NNV + i) * NNV + j] + diag;
          float v = e * a;
          sf[((size_t)b * NP + i) * NP + j] = v;
          atomicAdd(&rowsum[b * NP + i], v);
        }
      }
    }
  }
}

// ---------------- small elementwise kernels ----------------

__global__ void gcn_wt_transpose(const float* __restrict__ W,
                                 bf16* __restrict__ Wt, int nmat) {
  int idx = blockIdx.x * blockDim.x + threadIdx.x;
  if (idx >= nmat * DD * DD) return;
  int mat = idx >> 16, rem = idx & 65535, n = rem >> 8, k = rem & 255;
  Wt[idx] = f2bf(W[(size_t)mat * DD * DD + (size_t)k * DD + n]);
}

__global__ void gcn_init_x(const float* __restrict__ a,
                           float* __restrict__ xf, bf16* __restrict__ xb) {
  int idx = blockIdx.x * blockDim.x + threadIdx.x;
  if (idx >= BB * NP * DD) return;
  int d = idx & 255, i = (idx >> 8) & 1023, b = idx >> 18;
  float v = (i < NNV) ? a[((size_t)b * NNV + i) * DD + d] : 0.0f;
  xf[idx] = v;
  xb[idx] = f2bf(v);
}

__global__ void gcn_transpose_x(const bf16* __restrict__ xb,
                                bf16* __restrict__ xT) {
  int idx = blockIdx.x * blockDim.x + threadIdx.x;
  if (idx >= BB * DD * NP) return;
  int j = idx & 1023, d = (idx >> 10) & 255, b = idx >> 18;
  xT[idx] = xb[((size_t)b * NP + j) * DD + d];
}

__global__ void gcn_zero_f32(float* __restrict__ p, int n) {
  int i = blockIdx.x * blockDim.x + threadIdx.x;
  if (i < n) p[i] = 0.0f;
}

__global__ void gcn_normalize(const float* __restrict__ sf,
                              const float* __restrict__ rowsum,
                              bf16* __restrict__ Sb) {
  size_t idx = (size_t)blockIdx.x * blockDim.x + threadIdx.x;
  if (idx >= (size_t)BB * NP * NP) return;
  int j = (int)(idx & 1023), i = (int)((idx >> 10) & 1023), b = (int)(idx >> 20);
  float v = 0.0f;
  if (i < NNV && j < NNV) v = sf[idx] / rowsum[b * NP + i];
  Sb[idx] = f2bf(v);
}

__global__ void gcn_copy_out(const float* __restrict__ xf,
                             float* __restrict__ out) {
  int idx = blockIdx.x * blockDim.x + threadIdx.x;
  if (idx >= BB * NNV * DD) return;
  int d = idx & 255, rest = idx >> 8;
  int i = rest % NNV, b = rest / NNV;
  out[idx] = xf[((size_t)b * NP + i) * DD + d];
}

// ---------------- host orchestration ----------------

extern "C" void kernel_launch(void* const* d_in, const int* in_sizes, int n_in,
                              void* d_out, int out_size, void* d_ws, size_t ws_size,
                              hipStream_t stream) {
  (void)in_sizes; (void)n_in; (void)out_size; (void)ws_size;
  const float* amino    = (const float*)d_in[0];
  const float* contacts = (const float*)d_in[1];
  const float* W_list   = (const float*)d_in[2];
  const float* dense_W  = (const float*)d_in[3];
  const float* dense_b  = (const float*)d_in[4];
  const float* final_W  = (const float*)d_in[5];
  const float* final_b  = (const float*)d_in[6];
  float* out = (float*)d_out;

  char* ws = (char*)d_ws;
  size_t off = 0;
  auto carve = [&](size_t bytes) -> void* {
    void* p = ws + off;
    off += (bytes + 255) & ~(size_t)255;
    return p;
  };

  float* xf     = (float*)carve((size_t)BB * NP * DD * 4);
  bf16*  xb     = (bf16*) carve((size_t)BB * NP * DD * 2);
  bf16*  xT     = (bf16*) carve((size_t)BB * DD * NP * 2);
  bf16*  Yb     = (bf16*) carve((size_t)BB * NP * DD * 2);
  float* sf     = (float*)carve((size_t)BB * NP * NP * 4);
  bf16*  Sb     = (bf16*) carve((size_t)BB * NP * NP * 2);
  float* rowsum = (float*)carve((size_t)BB * NP * 4);
  bf16*  T1     = (bf16*) carve((size_t)BB * NP * DD * 2);
  bf16*  T2     = (bf16*) carve((size_t)BB * NP * DD * 2);
  float* ff     = (float*)carve((size_t)BB * NP * DD * 4);
  bf16*  WlT    = (bf16*) carve((size_t)LL * DD * DD * 2);
  bf16*  DWt    = (bf16*) carve((size_t)LL * NDLC * DD * DD * 2);
  bf16*  FWt    = (bf16*) carve((size_t)DD * DD * 2);

  const dim3 blk(256);
  const int M = BB * NP;                          // 16384 padded rows
  const int gemm_waves = (M >> 5) * (DD >> 6);    // 32x64 strips -> 2048 waves
  const int gemm_blocks = (gemm_waves * 32 + 255) / 256;

  // 0) weights -> bf16 transposed (B^T form)
  gcn_wt_transpose<<<(LL * DD * DD + 255) / 256, blk, 0, stream>>>(W_list, WlT, LL);
  gcn_wt_transpose<<<(LL * NDLC * DD * DD + 255) / 256, blk, 0, stream>>>(dense_W, DWt, LL * NDLC);
  gcn_wt_transpose<<<(DD * DD + 255) / 256, blk, 0, stream>>>(final_W, FWt, 1);
  // x init (zero-padded)
  gcn_init_x<<<(BB * NP * DD + 255) / 256, blk, 0, stream>>>(amino, xf, xb);

  for (int l = 0; l < LL; ++l) {
    // Y = x0 @ W_l  -> bf16
    gcn_wmma_gemm<<<gemm_blocks, blk, 0, stream>>>(
        xb, WlT + (size_t)l * DD * DD, nullptr, Yb,
        M, DD, DD, DD, DD, DD, 0, 0, nullptr, 0, nullptr);

    // xT for aggregation B^T
    gcn_transpose_x<<<(BB * DD * NP + 255) / 256, blk, 0, stream>>>(xb, xT);

    // rowsum = 0
    gcn_zero_f32<<<(BB * NP + 255) / 256, blk, 0, stream>>>(rowsum, BB * NP);

    // scores + exp + mask + rowsum
    {
      int waves = BB * (NP / 32) * (NP / 64);     // 8192
      gcn_scores<<<(waves * 32 + 255) / 256, blk, 0, stream>>>(
          Yb, xb, contacts, sf, rowsum);
    }

    // normalize -> bf16 (zero pads)
    {
      size_t tot = (size_t)BB * NP * NP;
      gcn_normalize<<<(int)((tot + 255) / 256), blk, 0, stream>>>(sf, rowsum, Sb);
    }

    // x = s @ x0   (A = Sb, B^T = xT per batch)
    gcn_wmma_gemm<<<gemm_blocks, blk, 0, stream>>>(
        Sb, xT, nullptr, T1,
        M, DD, NP, NP, NP, DD,
        (size_t)DD * NP, NP, nullptr, 0, nullptr);

    // dense 0: relu(x @ W + b)
    gcn_wmma_gemm<<<gemm_blocks, blk, 0, stream>>>(
        T1, DWt + (size_t)(l * NDLC + 0) * DD * DD, nullptr, T2,
        M, DD, DD, DD, DD, DD, 0, 0,
        dense_b + (size_t)(l * NDLC + 0) * DD, 1, nullptr);

    // dense 1 + skip: x = relu(x @ W + b) + x0  (in-place update of xf/xb)
    gcn_wmma_gemm<<<gemm_blocks, blk, 0, stream>>>(
        T2, DWt + (size_t)(l * NDLC + 1) * DD * DD, xf, xb,
        M, DD, DD, DD, DD, DD, 0, 0,
        dense_b + (size_t)(l * NDLC + 1) * DD, 1, xf);
  }

  // final: out = x @ final_W + final_b
  gcn_wmma_gemm<<<gemm_blocks, blk, 0, stream>>>(
      xb, FWt, ff, nullptr,
      M, DD, DD, DD, DD, DD, 0, 0, final_b, 0, nullptr);

  gcn_copy_out<<<(BB * NNV * DD + 255) / 256, blk, 0, stream>>>(ff, out);
}
